// RecursiveLSTM_49228915147484
// MI455X (gfx1250) — matrix operand order repaired and checked by reading, not compiled, not run on hardware
//
#include <hip/hip_runtime.h>
#include <hip/hip_bf16.h>
#include <stdint.h>

typedef __attribute__((ext_vector_type(16))) __bf16 v16bf;
typedef __attribute__((ext_vector_type(8)))  __bf16 v8bf;
typedef __attribute__((ext_vector_type(8)))  float  v8f;

#define B_BATCH 32
#define T_STEPS 64
#define T_CHUNK 16            // logits ring-buffer depth (64B output bursts)
#define E_DIM   512
#define H_DIM   1024
#define G_DIM   (4 * H_DIM)   // 4096
#define V_DIM   32000

// ---------- helpers ----------
static __device__ __forceinline__ uint16_t f32_to_bf16(float f) {
    union { float f; uint32_t u; } x; x.f = f;
    uint32_t u = x.u;
    u += 0x7FFFu + ((u >> 16) & 1u);   // round-to-nearest-even
    return (uint16_t)(u >> 16);
}

// ---------- one-time: fp32 -> bf16 weight conversion (grid-stride) ----------
__global__ void cvt_f32_bf16_kernel(const float* __restrict__ src,
                                    uint16_t* __restrict__ dst, int n) {
    int i = blockIdx.x * blockDim.x + threadIdx.x;
    int stride = gridDim.x * blockDim.x;
    for (; i < n; i += stride) dst[i] = f32_to_bf16(src[i]);
}

// ---------- one-time: h0 = cond@Wh.T+bh (bf16), c0 = cond@Wc.T+bc (f32), tok=START ----------
__global__ void init_state_kernel(const float* __restrict__ cond,
                                  const float* __restrict__ Wh, const float* __restrict__ bh,
                                  const float* __restrict__ Wc, const float* __restrict__ bc,
                                  float* __restrict__ c_state,
                                  uint16_t* __restrict__ h_bf,
                                  int* __restrict__ tok) {
    int idx = blockIdx.x * blockDim.x + threadIdx.x;      // 0 .. 2*32*1024-1
    int half = idx / (B_BATCH * H_DIM);
    int r    = idx % (B_BATCH * H_DIM);
    int b = r / H_DIM, j = r % H_DIM;
    const float* W    = half ? Wc : Wh;
    const float* bias = half ? bc : bh;
    float s = bias[j];
    const float* cb = cond + (size_t)b * E_DIM;
    const float* wr = W    + (size_t)j * E_DIM;
    #pragma unroll 4
    for (int k = 0; k < E_DIM; ++k) s += cb[k] * wr[k];
    if (half) c_state[r] = s;
    else      h_bf[r] = f32_to_bf16(s);
    if (idx < B_BATCH) tok[idx] = 1;                      // START_TOK
}

// ---------- per-step: x = bf16(embed[tok]) ----------
__global__ void gather_embed_kernel(const float* __restrict__ table,
                                    const int* __restrict__ tok,
                                    uint16_t* __restrict__ x_bf) {
    int i = blockIdx.x * blockDim.x + threadIdx.x;        // 32*512
    int b = i / E_DIM, k = i % E_DIM;
    x_bf[i] = f32_to_bf16(table[(size_t)tok[b] * E_DIM + k]);
}

// ---------- WMMA GEMM: out[32,N] (f32) (+)= A[32,K](bf16) x W[N,K](bf16)^T + bias ----------
// One wave per 16x16 output tile; 8 waves per block.
// A layout (matches B): lane<16 -> row=lane, K=k0..k0+15
//                      lane>=16 -> row=lane-16, K=k0+16..k0+31
// B layout: lane<16 -> col=lane, K=k0..k0+15 ; lane>=16 -> col=lane-16, K=k0+16..k0+31
//   (W row-major [N,K] => contiguous 32B load per lane)
__global__ void __launch_bounds__(256)
wmma_gemm_kernel(float* __restrict__ out,
                 const uint16_t* __restrict__ A,
                 const uint16_t* __restrict__ W,
                 const float* __restrict__ bias0,
                 const float* __restrict__ bias1,
                 int N, int K, int accumulate) {
    const int lane = threadIdx.x & 31;
    const int wave = threadIdx.x >> 5;
    const int tile = blockIdx.x * 8 + wave;
    const int ntiles_n = N >> 4;
    const int mt = tile / ntiles_n;
    const int nt = tile % ntiles_n;
    const int half = lane >> 4;
    const int l15  = lane & 15;
    const int mrow = mt * 16 + l15;
    const int ncol = nt * 16 + l15;
    const uint16_t* __restrict__ Arow = A + (size_t)mrow * K;
    const uint16_t* __restrict__ Wrow = W + (size_t)ncol * K;

    v8f acc = {};
    for (int k0 = 0; k0 < K; k0 += 32) {
        union { v16bf v; v8bf h[2]; } a;
        a.h[0] = *(const v8bf*)(Arow + k0 + half * 16);
        a.h[1] = *(const v8bf*)(Arow + k0 + half * 16 + 8);
        v16bf bmat = *(const v16bf*)(Wrow + k0 + half * 16);
        __builtin_prefetch(Wrow + k0 + 128, 0, 1);   // global_prefetch_b8: next weight chunk
        acc = __builtin_amdgcn_wmma_f32_16x16x32_bf16(
                  /*neg_a=*/false, a.v, /*neg_b=*/false, bmat,
                  /*c_mod=*/(short)0, acc, /*reuse_a=*/false, /*reuse_b=*/false);
    }

    float badd = 0.0f;
    if (bias0) badd += bias0[ncol];
    if (bias1) badd += bias1[ncol];
    #pragma unroll
    for (int r = 0; r < 8; ++r) {
        int m = mt * 16 + r + 8 * half;                 // D: VGPR r -> M=r (+8 for hi lanes)
        size_t o = (size_t)m * N + ncol;                // N = l15 within tile
        float v = acc[r] + badd;
        if (accumulate) out[o] += v; else out[o] = v;
    }
}

// ---------- per-step: LSTM cell pointwise ----------
__global__ void lstm_cell_kernel(const float* __restrict__ gates,
                                 float* __restrict__ c_state,
                                 uint16_t* __restrict__ h_bf) {
    int i = blockIdx.x * blockDim.x + threadIdx.x;      // 32*1024
    int b = i / H_DIM, j = i % H_DIM;
    const float* gr = gates + (size_t)b * G_DIM;
    float ig = gr[j];
    float fg = gr[H_DIM + j];
    float gg = gr[2 * H_DIM + j];
    float og = gr[3 * H_DIM + j];
    float is = 1.0f / (1.0f + __expf(-ig));
    float fs = 1.0f / (1.0f + __expf(-fg));
    float gt = tanhf(gg);
    float os = 1.0f / (1.0f + __expf(-og));
    float cn = fs * c_state[i] + is * gt;
    c_state[i] = cn;
    h_bf[i] = f32_to_bf16(os * tanhf(cn));
}

// ---------- per-step: Gumbel-max sample from contiguous logits slab ----------
__global__ void __launch_bounds__(256)
sample_kernel(const float* __restrict__ logits,    // [32, V] (ring slab for step t)
              int* __restrict__ tok,
              float* __restrict__ out_caps,        // [32, T]
              int t) {
    const int b = blockIdx.x;
    const int tid = threadIdx.x;
    const float* lr = logits + (size_t)b * V_DIM;
    float best = -1e30f; int bestv = 0;
    for (int v = tid; v < V_DIM; v += 256) {
        float lv = lr[v];
        // deterministic hash -> uniform -> gumbel
        uint32_t s = (uint32_t)v * 2654435761u ^ ((uint32_t)b * 40503u + 13u)
                   ^ ((uint32_t)t * 1013904223u);
        s ^= s >> 16; s *= 2246822519u; s ^= s >> 13; s *= 3266489917u; s ^= s >> 16;
        float u = (float)(s >> 8) * (1.0f / 16777216.0f) + 1e-12f;
        float val = lv - __logf(-__logf(u));
        if (val > best) { best = val; bestv = v; }
    }
    __shared__ float smax[256];
    __shared__ int   sidx[256];
    smax[tid] = best; sidx[tid] = bestv;
    __syncthreads();
    for (int off = 128; off > 0; off >>= 1) {
        if (tid < off && smax[tid + off] > smax[tid]) {
            smax[tid] = smax[tid + off]; sidx[tid] = sidx[tid + off];
        }
        __syncthreads();
    }
    if (tid == 0) {
        tok[b] = sidx[0];
        out_caps[b * T_STEPS + t] = (float)sidx[0];
    }
}

// ---------- every T_CHUNK steps: transpose [tc][B*V] ring buffer -> [B,V,T] output ----------
// Reads are fully coalesced (consecutive threads read consecutive addresses per slab).
// Each thread writes one 64B contiguous, sector-aligned burst -> no HBM write amplification.
__global__ void __launch_bounds__(256)
transpose_logits_kernel(const float* __restrict__ buf,   // [T_CHUNK][B*V]
                        float* __restrict__ out_logits,  // [B*V][T]
                        int t0) {
    int i = blockIdx.x * blockDim.x + threadIdx.x;       // 0 .. B*V-1  (i = b*V + v)
    float vals[T_CHUNK];
    #pragma unroll
    for (int tc = 0; tc < T_CHUNK; ++tc)
        vals[tc] = buf[(size_t)tc * (B_BATCH * V_DIM) + i];
    float* dst = out_logits + (size_t)i * T_STEPS + t0;  // 64B-aligned (t0 % 16 == 0)
    #pragma unroll
    for (int q = 0; q < T_CHUNK / 4; ++q) {
        float4 w = make_float4(vals[4*q], vals[4*q+1], vals[4*q+2], vals[4*q+3]);
        ((float4*)dst)[q] = w;
    }
}

// ---------- launcher ----------
extern "C" void kernel_launch(void* const* d_in, const int* in_sizes, int n_in,
                              void* d_out, int out_size, void* d_ws, size_t ws_size,
                              hipStream_t stream) {
    (void)in_sizes; (void)n_in; (void)out_size; (void)ws_size;
    const float* condition = (const float*)d_in[0];
    const float* Wh    = (const float*)d_in[1];
    const float* bh    = (const float*)d_in[2];
    const float* Wc    = (const float*)d_in[3];
    const float* bc    = (const float*)d_in[4];
    const float* embed = (const float*)d_in[5];
    const float* W_ih  = (const float*)d_in[6];
    const float* b_ih  = (const float*)d_in[7];
    const float* W_hh  = (const float*)d_in[8];
    const float* b_hh  = (const float*)d_in[9];
    const float* W_out = (const float*)d_in[10];
    const float* b_out = (const float*)d_in[11];

    float* out_caps   = (float*)d_out;                     // [32,64]
    float* out_logits = out_caps + B_BATCH * T_STEPS;      // [32,32000,64]

    // workspace carve (256B aligned)
    char* ws = (char*)d_ws;
    size_t off = 0;
    auto carve = [&](size_t bytes) -> char* {
        char* p = ws + off; off += (bytes + 255) & ~(size_t)255; return p;
    };
    uint16_t* Wih_bf  = (uint16_t*)carve((size_t)G_DIM * E_DIM * 2);    // 4 MB
    uint16_t* Whh_bf  = (uint16_t*)carve((size_t)G_DIM * H_DIM * 2);    // 8 MB
    uint16_t* Wout_bf = (uint16_t*)carve((size_t)V_DIM * H_DIM * 2);    // 64 MB (L2-resident)
    uint16_t* x_bf    = (uint16_t*)carve((size_t)B_BATCH * E_DIM * 2);
    uint16_t* h_bf    = (uint16_t*)carve((size_t)B_BATCH * H_DIM * 2);
    float*    gates   = (float*)carve((size_t)B_BATCH * G_DIM * 4);
    float*    c_state = (float*)carve((size_t)B_BATCH * H_DIM * 4);
    float*    log_buf = (float*)carve((size_t)T_CHUNK * B_BATCH * V_DIM * 4); // 64 MB ring
    int*      tok     = (int*)carve(B_BATCH * 4);

    // one-time: bf16 weight images in workspace (recomputed every call; deterministic)
    cvt_f32_bf16_kernel<<<1024, 256, 0, stream>>>(W_ih,  Wih_bf,  G_DIM * E_DIM);
    cvt_f32_bf16_kernel<<<1024, 256, 0, stream>>>(W_hh,  Whh_bf,  G_DIM * H_DIM);
    cvt_f32_bf16_kernel<<<2048, 256, 0, stream>>>(W_out, Wout_bf, V_DIM * H_DIM);

    // one-time: initial state
    init_state_kernel<<<(2 * B_BATCH * H_DIM) / 256, 256, 0, stream>>>(
        condition, Wh, bh, Wc, bc, c_state, h_bf, tok);

    const int gates_blocks  = (2 * (G_DIM / 16)) / 8;   // 64
    const int logits_blocks = (2 * (V_DIM / 16)) / 8;   // 500

    for (int t = 0; t < T_STEPS; ++t) {
        float* log_slab = log_buf + (size_t)(t % T_CHUNK) * (B_BATCH * V_DIM);

        gather_embed_kernel<<<(B_BATCH * E_DIM) / 256, 256, 0, stream>>>(embed, tok, x_bf);
        // gates = x @ W_ih^T + (b_ih + b_hh)
        wmma_gemm_kernel<<<gates_blocks, 256, 0, stream>>>(
            gates, x_bf, Wih_bf, b_ih, b_hh, G_DIM, E_DIM, 0);
        // gates += h @ W_hh^T
        wmma_gemm_kernel<<<gates_blocks, 256, 0, stream>>>(
            gates, h_bf, Whh_bf, nullptr, nullptr, G_DIM, H_DIM, 1);
        lstm_cell_kernel<<<(B_BATCH * H_DIM) / 256, 256, 0, stream>>>(gates, c_state, h_bf);
        // logits = h_new @ W_out^T + b_out  (contiguous slab in ring buffer)
        wmma_gemm_kernel<<<logits_blocks, 256, 0, stream>>>(
            log_slab, h_bf, Wout_bf, b_out, nullptr, V_DIM, H_DIM, 0);
        sample_kernel<<<B_BATCH, 256, 0, stream>>>(log_slab, tok, out_caps, t);

        if ((t % T_CHUNK) == T_CHUNK - 1) {
            transpose_logits_kernel<<<(B_BATCH * V_DIM) / 256, 256, 0, stream>>>(
                log_buf, out_logits, t - (T_CHUNK - 1));
        }
    }
}
